// ScaledDotProductAttention_69587060129832
// MI455X (gfx1250) — matrix operand order repaired and checked by reading, not compiled, not run on hardware
//
#include <hip/hip_runtime.h>

typedef __attribute__((ext_vector_type(16))) _Float16 v16h;
typedef __attribute__((ext_vector_type(8)))  float    v8f;

constexpr int kB = 8;
constexpr int kS = 2048;
constexpr int kD = 256;
constexpr int ROWS = 16;              // q rows per workgroup
constexpr int WAVES = 16;
constexpr int THREADS = WAVES * 32;   // wave32
constexpr float SCALE = 0.0625f;      // 1/sqrt(256)
constexpr int VKEYS = 64;             // V keys staged per phase-3 step

// LDS layout (dynamic shared):
//   sc   : 16 x 2048 f32 scores/probs        = 131072 B
//   vt   : 256 x 64 f16 (V tile, [d][k])     =  32768 B
//   part : 512 f32 reduction partials        =   2048 B
//   rowv : 16 f32 row max/sum                =     64 B
constexpr size_t SC_BYTES = (size_t)ROWS * kS * 4;
constexpr size_t VT_BYTES = (size_t)kD * VKEYS * 2;
constexpr size_t SMEM_BYTES = SC_BYTES + VT_BYTES + (THREADS + ROWS) * 4;

__global__ __launch_bounds__(THREADS)
void sdpa_wmma_kernel(const float* __restrict__ qg,
                      const float* __restrict__ kg,
                      const float* __restrict__ vg,
                      const int*   __restrict__ maskg,
                      float* __restrict__ outg,
                      float* __restrict__ attng)
{
    extern __shared__ char smem[];
    float*    sc   = (float*)smem;
    _Float16* vt   = (_Float16*)(smem + SC_BYTES);
    float*    part = (float*)(smem + SC_BYTES + VT_BYTES);
    float*    rowv = part + THREADS;

    const int tid  = threadIdx.x;
    const int wave = tid >> 5;
    const int lane = tid & 31;
    const int lhi  = lane >> 4;   // lane-half select
    const int llo  = lane & 15;
    const int koff = lhi * 16;    // A-fragment K offset per lane half

    const int bidx = blockIdx.x / (kS / ROWS);
    const int q0   = (blockIdx.x % (kS / ROWS)) * ROWS;

    // Prefetch this workgroup's mask slab (16 rows x 8 KB = 128 KB) so the
    // HBM mask stream overlaps the QK^T compute phase.
    {
        const char* mbase = (const char*)(maskg + ((size_t)bidx * kS + q0) * kS);
        __builtin_prefetch(mbase + (size_t)tid * 256,       0, 1);
        __builtin_prefetch(mbase + (size_t)tid * 256 + 128, 0, 1);
    }

    // ---------------- Phase 1: scores = scale * Q K^T ----------------
    // A fragment rows: m = llo; wave owns keys [wave*128, wave*128+128)
    const float* qrow = qg + ((size_t)bidx * kS + q0 + llo) * kD;
    const int key00 = wave * 128;

    v8f acc[8];
    #pragma unroll
    for (int t = 0; t < 8; ++t)
        acc[t] = (v8f){0.f,0.f,0.f,0.f,0.f,0.f,0.f,0.f};

    #pragma unroll 1
    for (int dc = 0; dc < kD; dc += 32) {
        // A fragment: a[j] = Q[m][dc + koff + j]
        v16h aq;
        {
            const float* p0 = qrow + dc + koff;
            #pragma unroll
            for (int j = 0; j < 8; ++j) aq[j]     = (_Float16)p0[j];
            #pragma unroll
            for (int j = 0; j < 8; ++j) aq[8 + j] = (_Float16)p0[8 + j];
        }
        // Build 4 B fragments, THEN issue their 4 WMMAs. All four fragments
        // are live at the first WMMA -> distinct registers -> the WAR hazard
        // (WMMA source overwritten by next v_cvt) is pushed >=3 WMMAs back,
        // eliminating the 4x v_nop per WMMA seen with a recycled register set.
        #pragma unroll
        for (int tg = 0; tg < 8; tg += 4) {
            v16h bk[4];
            #pragma unroll
            for (int u = 0; u < 4; ++u) {
                const int key = key00 + (tg + u) * 16 + llo;   // B col n = llo
                const float* kp = kg + ((size_t)bidx * kS + key) * kD + dc + lhi * 16;
                #pragma unroll
                for (int j = 0; j < 16; ++j) bk[u][j] = (_Float16)kp[j];
            }
            #pragma unroll
            for (int u = 0; u < 4; ++u) {
                acc[tg + u] = __builtin_amdgcn_wmma_f32_16x16x32_f16(
                    false, aq, false, bk[u], (short)0, acc[tg + u], false, false);
            }
        }
    }

    // Scatter scaled scores into LDS: C layout M = v + 8*lhi, N = llo
    #pragma unroll
    for (int t = 0; t < 8; ++t) {
        const int col = key00 + t * 16 + llo;
        #pragma unroll
        for (int vv = 0; vv < 8; ++vv) {
            const int row = vv + 8 * lhi;
            sc[row * kS + col] = acc[t][vv] * SCALE;
        }
    }
    __syncthreads();

    // ---------------- Phase 2: mask + softmax + write attn ----------------
    {
        const int r   = wave;        // row 0..15
        const int seg = lane;        // 32 segments of 64 columns
        const int base = seg * 64;
        const int*   mrow = maskg + ((size_t)bidx * kS + q0 + r) * kS + base;
        float*       srow = sc + r * kS + base;

        float mx = -3.0e38f;
        #pragma unroll 4
        for (int j = 0; j < 64; ++j) {
            float s = srow[j];
            if (mrow[j] == 0) s = -1.0e9f;
            srow[j] = s;
            mx = fmaxf(mx, s);
        }
        part[tid] = mx;
        __syncthreads();
        if (tid < ROWS) {
            float m2 = -3.0e38f;
            for (int i = 0; i < 32; ++i) m2 = fmaxf(m2, part[tid * 32 + i]);
            rowv[tid] = m2;
        }
        __syncthreads();

        const float rmax = rowv[r];
        float sum = 0.f;
        #pragma unroll 4
        for (int j = 0; j < 64; ++j) {
            float e = __expf(srow[j] - rmax);
            srow[j] = e;
            sum += e;
        }
        part[tid] = sum;
        __syncthreads();
        if (tid < ROWS) {
            float s2 = 0.f;
            for (int i = 0; i < 32; ++i) s2 += part[tid * 32 + i];
            rowv[tid] = s2;
        }
        __syncthreads();

        const float inv = 1.0f / rowv[r];
        float* arow = attng + ((size_t)bidx * kS + q0 + r) * kS + base;
        #pragma unroll 4
        for (int j = 0; j < 64; ++j) {
            float p = srow[j] * inv;
            srow[j] = p;
            arow[j] = p;
        }
    }
    __syncthreads();

    // ---------------- Phase 3: out = P V ----------------
    // Wave owns D columns [wave*16, wave*16+16); stage 64 keys per step
    // (2 WMMAs per barrier pair).
    v8f oacc = (v8f){0.f,0.f,0.f,0.f,0.f,0.f,0.f,0.f};
    const int dcol = wave * 16 + llo;

    #pragma unroll 1
    for (int kc = 0; kc < kS; kc += VKEYS) {
        __syncthreads();   // previous V tile fully consumed
        // Stage V[kc..kc+64) x 256 into LDS transposed f16: vt[d*64 + kr]
        {
            const int vr    = tid >> 3;         // 0..63 key row
            const int cbase = (tid & 7) * 32;   // 32-col segment
            const float* vp = vg + ((size_t)bidx * kS + kc + vr) * kD + cbase;
            #pragma unroll
            for (int c = 0; c < 32; ++c)
                vt[(cbase + c) * VKEYS + vr] = (_Float16)vp[c];
        }
        __syncthreads();

        // Build BOTH fragment pairs first, then issue both WMMAs, so the
        // fragments occupy distinct registers (no WMMA->v_cvt WAR nops).
        v16h pa[2], bv[2];
        #pragma unroll
        for (int s = 0; s < 2; ++s) {
            const float* prow = sc + llo * kS + kc + s * 32 + koff;
            #pragma unroll
            for (int j = 0; j < 8; ++j) pa[s][j]     = (_Float16)prow[j];
            #pragma unroll
            for (int j = 0; j < 8; ++j) pa[s][8 + j] = (_Float16)prow[8 + j];
            const _Float16* bp = vt + dcol * VKEYS + s * 32 + lhi * 16;
            #pragma unroll
            for (int j = 0; j < 16; ++j) bv[s][j] = bp[j];
        }
        #pragma unroll
        for (int s = 0; s < 2; ++s) {
            oacc = __builtin_amdgcn_wmma_f32_16x16x32_f16(
                false, pa[s], false, bv[s], (short)0, oacc, false, false);
        }
    }

    // Store out tile: M = v + 8*lhi, N = dcol
    float* op = outg + ((size_t)bidx * kS + q0) * kD + dcol;
    #pragma unroll
    for (int vv = 0; vv < 8; ++vv) {
        const int row = vv + 8 * lhi;
        op[(size_t)row * kD] = oacc[vv];
    }
}

extern "C" void kernel_launch(void* const* d_in, const int* in_sizes, int n_in,
                              void* d_out, int out_size, void* d_ws, size_t ws_size,
                              hipStream_t stream) {
    (void)in_sizes; (void)n_in; (void)out_size; (void)d_ws; (void)ws_size;
    const float* q    = (const float*)d_in[0];
    const float* k    = (const float*)d_in[1];
    const float* v    = (const float*)d_in[2];
    const int*   mask = (const int*)d_in[3];

    float* out  = (float*)d_out;                       // [B,S,D]
    float* attn = out + (size_t)kB * kS * kD;          // [B,S,S] follows

    (void)hipFuncSetAttribute((const void*)sdpa_wmma_kernel,
                              hipFuncAttributeMaxDynamicSharedMemorySize,
                              (int)SMEM_BYTES);

    const int grid = kB * (kS / ROWS);                 // 1024 workgroups
    sdpa_wmma_kernel<<<grid, THREADS, SMEM_BYTES, stream>>>(q, k, v, mask, out, attn);
}
